// _LTX2Attention_27590869909702
// MI455X (gfx1250) — matrix operand order repaired and checked by reading, not compiled, not run on hardware
//
#include <hip/hip_runtime.h>

// ---------------------------------------------------------------------------
// LTX2 attention block for MI455X (gfx1250, wave32, WMMA).
// All matmuls in bf16 WMMA (v_wmma_f32_16x16x32_bf16), f32 accumulate.
// ---------------------------------------------------------------------------

typedef __bf16 bf16_t;
typedef __bf16 v16bf __attribute__((ext_vector_type(16)));
typedef __bf16 v8bf  __attribute__((ext_vector_type(8)));
typedef __bf16 v4bf  __attribute__((ext_vector_type(4)));
typedef float  v8f   __attribute__((ext_vector_type(8)));
typedef float  v4f   __attribute__((ext_vector_type(4)));

union V16 { v16bf v; v8bf h[2]; };

__device__ __forceinline__ v8f wmma_bf16(v16bf a, v16bf b, v8f c) {
  return __builtin_amdgcn_wmma_f32_16x16x32_bf16(false, a, false, b,
                                                 (short)0, c, false, false);
}

#define S_LEN   2048
#define D_MODEL 2048
#define NHEAD   32
#define DHEAD   64

// ---------------------------------------------------------------------------
// fp32 -> bf16 elementwise convert (4 elems / thread)
// ---------------------------------------------------------------------------
__global__ __launch_bounds__(256)
void k_cvt_bf16(const float* __restrict__ in, bf16_t* __restrict__ out, int n4) {
  int i = blockIdx.x * 256 + threadIdx.x;
  if (i < n4) {
    v4f x = *(const v4f*)(in + (size_t)i * 4);
    v4bf y;
#pragma unroll
    for (int k = 0; k < 4; k++) y[k] = (bf16_t)x[k];
    *(v4bf*)(out + (size_t)i * 4) = y;
  }
}

// ---------------------------------------------------------------------------
// W[K][N] fp32 -> Wt[N][K] bf16  (LDS-tiled transpose, 32x32 tiles)
// ---------------------------------------------------------------------------
__global__ __launch_bounds__(256)
void k_transpose_cvt(const float* __restrict__ W, bf16_t* __restrict__ Wt) {
  __shared__ float tile[32][33];
  const int tx = threadIdx.x, ty = threadIdx.y;
  const int n = blockIdx.x * 32 + tx;
  const int k = blockIdx.y * 32 + ty;
#pragma unroll
  for (int i = 0; i < 32; i += 8)
    tile[ty + i][tx] = W[(size_t)(k + i) * D_MODEL + n];
  __syncthreads();
  const int ko = blockIdx.y * 32 + tx;
  const int no = blockIdx.x * 32 + ty;
#pragma unroll
  for (int i = 0; i < 32; i += 8)
    Wt[(size_t)(no + i) * D_MODEL + ko] = (bf16_t)tile[tx][ty + i];
}

// ---------------------------------------------------------------------------
// GEMM: C[M][N] = A[M][K] * Bt[N][K]^T + bias[N], bf16 in, fp32 out.
// 256 threads = 8 waves arranged 4(M) x 2(N); wave tile 32x64 (2x4 WMMA
// accumulators); block tile 128x128.  Per 32-K step: 12 b128 loads feed
// 8 WMMAs (A operands reused across 4 B tiles).
// A operand: k = 16*half + s -> per-lane chunks at half*16 / half*16+8.
// B operand: K-striped layout    -> contiguous 16 elems at half*16.
// C/D: row = r + 8*half, col = lane&15.
// ---------------------------------------------------------------------------
__global__ __launch_bounds__(256)
void k_gemm_bf16(const bf16_t* __restrict__ A, const bf16_t* __restrict__ Bt,
                 const float* __restrict__ bias, float* __restrict__ C,
                 int M, int N, int K) {
  const int lane = threadIdx.x & 31;
  const int wave = threadIdx.x >> 5;
  const int half = lane >> 4, l15 = lane & 15;
  const int m0 = blockIdx.y * 128 + (wave >> 1) * 32;
  const int n0 = blockIdx.x * 128 + (wave & 1) * 64;

  v8f acc[2][4] = {};
  const bf16_t* a0p = A + (size_t)(m0 + l15) * K;
  const bf16_t* a1p = a0p + (size_t)16 * K;
  const bf16_t* bp0 = Bt + (size_t)(n0 + l15) * K;
  const bf16_t* bp1 = bp0 + (size_t)16 * K;
  const bf16_t* bp2 = bp0 + (size_t)32 * K;
  const bf16_t* bp3 = bp0 + (size_t)48 * K;

  for (int k0 = 0; k0 < K; k0 += 32) {
    const int ka = k0 + half * 16;
    V16 a0, a1;
    a0.h[0] = *(const v8bf*)(a0p + ka);
    a0.h[1] = *(const v8bf*)(a0p + ka + 8);
    a1.h[0] = *(const v8bf*)(a1p + ka);
    a1.h[1] = *(const v8bf*)(a1p + ka + 8);
    const int kb = k0 + half * 16;
    v16bf b0 = *(const v16bf*)(bp0 + kb);
    v16bf b1 = *(const v16bf*)(bp1 + kb);
    v16bf b2 = *(const v16bf*)(bp2 + kb);
    v16bf b3 = *(const v16bf*)(bp3 + kb);
    acc[0][0] = wmma_bf16(a0.v, b0, acc[0][0]);
    acc[0][1] = wmma_bf16(a0.v, b1, acc[0][1]);
    acc[0][2] = wmma_bf16(a0.v, b2, acc[0][2]);
    acc[0][3] = wmma_bf16(a0.v, b3, acc[0][3]);
    acc[1][0] = wmma_bf16(a1.v, b0, acc[1][0]);
    acc[1][1] = wmma_bf16(a1.v, b1, acc[1][1]);
    acc[1][2] = wmma_bf16(a1.v, b2, acc[1][2]);
    acc[1][3] = wmma_bf16(a1.v, b3, acc[1][3]);
  }

#pragma unroll
  for (int qi = 0; qi < 2; qi++)
#pragma unroll
    for (int ni = 0; ni < 4; ni++) {
      const int col = n0 + ni * 16 + l15;
      const float bv = bias[col];
#pragma unroll
      for (int r = 0; r < 8; r++) {
        const int row = m0 + qi * 16 + half * 8 + r;
        C[(size_t)row * N + col] = acc[qi][ni][r] + bv;
      }
    }
}

// ---------------------------------------------------------------------------
// RMSNorm(full D) * w, then interleaved RoPE, write head-split bf16:
// out[H][S][Dh].  One block (256 thr) per sequence row; 8 elems/thread.
// ---------------------------------------------------------------------------
__global__ __launch_bounds__(256)
void k_rmsnorm_rope(const float* __restrict__ X, const float* __restrict__ w,
                    const float* __restrict__ cosT, const float* __restrict__ sinT,
                    bf16_t* __restrict__ outH) {
  const int s = blockIdx.x;
  const int t = threadIdx.x;
  const int lane = t & 31, wave = t >> 5;
  const size_t rowoff = (size_t)s * D_MODEL;
  const int e = t * 8;

  v4f x0 = *(const v4f*)(X + rowoff + e);
  v4f x1 = *(const v4f*)(X + rowoff + e + 4);
  float xv[8];
#pragma unroll
  for (int i = 0; i < 4; i++) { xv[i] = x0[i]; xv[4 + i] = x1[i]; }

  float ss = 0.f;
#pragma unroll
  for (int i = 0; i < 8; i++) ss += xv[i] * xv[i];
#pragma unroll
  for (int mk = 16; mk >= 1; mk >>= 1) ss += __shfl_xor(ss, mk, 32);

  __shared__ float part[8];
  if (lane == 0) part[wave] = ss;
  __syncthreads();
  float tot = 0.f;
#pragma unroll
  for (int i = 0; i < 8; i++) tot += part[i];
  const float r = rsqrtf(tot * (1.0f / (float)D_MODEL) + 1e-6f);

  v4f c0 = *(const v4f*)(cosT + rowoff + e);
  v4f c1 = *(const v4f*)(cosT + rowoff + e + 4);
  v4f sn0 = *(const v4f*)(sinT + rowoff + e);
  v4f sn1 = *(const v4f*)(sinT + rowoff + e + 4);
  v4f w0 = *(const v4f*)(w + e);
  v4f w1 = *(const v4f*)(w + e + 4);

  float y[8], cc[8], sn[8];
#pragma unroll
  for (int i = 0; i < 4; i++) {
    y[i]     = xv[i]     * r * w0[i];
    y[4 + i] = xv[4 + i] * r * w1[i];
    cc[i] = c0[i]; cc[4 + i] = c1[i];
    sn[i] = sn0[i]; sn[4 + i] = sn1[i];
  }
  float o[8];
#pragma unroll
  for (int i = 0; i < 8; i += 2) {
    o[i]     = y[i]     * cc[i]     - y[i + 1] * sn[i];
    o[i + 1] = y[i + 1] * cc[i + 1] + y[i]     * sn[i + 1];
  }

  const int hh = e >> 6, dh = e & 63;
  bf16_t* dst = outH + ((size_t)hh * S_LEN + s) * DHEAD + dh;
  v8bf ov;
#pragma unroll
  for (int i = 0; i < 8; i++) ov[i] = (bf16_t)o[i];
  *(v8bf*)dst = ov;
}

// ---------------------------------------------------------------------------
// V[S][D] fp32 -> Vt[H][Dh][S] bf16 (transposed per head for PV WMMA B-operand)
// ---------------------------------------------------------------------------
__global__ __launch_bounds__(256)
void k_reorder_v(const float* __restrict__ V, bf16_t* __restrict__ Vt) {
  const size_t idx = (size_t)blockIdx.x * 256 + threadIdx.x;
  const int sidx = (int)(idx & (S_LEN - 1));
  const int hd   = (int)(idx >> 11);              // h*64 + dh
  Vt[idx] = (bf16_t)V[(size_t)sidx * D_MODEL + hd];
}

// ---------------------------------------------------------------------------
// Flash attention: grid (H, S/256); 8 waves/block, wave owns 32 queries.
// Streams keys in 32-tiles: 8 WMMAs QK^T + online softmax + 8 WMMAs PV.
// P goes C-layout -> LDS -> A-layout (in-wave transpose).
// ---------------------------------------------------------------------------
__global__ __launch_bounds__(256)
void k_attention(const bf16_t* __restrict__ Qh, const bf16_t* __restrict__ Kh,
                 const bf16_t* __restrict__ Vt, bf16_t* __restrict__ Out) {
  const int h = blockIdx.x;
  const int lane = threadIdx.x & 31, wave = threadIdx.x >> 5;
  const int half = lane >> 4, l15 = lane & 15;
  const int q0 = blockIdx.y * 256 + wave * 32;
  __shared__ __align__(16) bf16_t pst[8][32][40];   // per-wave 32x32 P staging

  const bf16_t* Qb = Qh + (size_t)h * S_LEN * DHEAD;
  const bf16_t* Kb = Kh + (size_t)h * S_LEN * DHEAD;
  const bf16_t* Vb = Vt + (size_t)h * DHEAD * S_LEN;

  V16 qa[2][2];                                      // [query-tile][k-step]
#pragma unroll
  for (int qi = 0; qi < 2; qi++) {
    const bf16_t* qr = Qb + (size_t)(q0 + qi * 16 + l15) * DHEAD;
#pragma unroll
    for (int kk = 0; kk < 2; kk++) {
      qa[qi][kk].h[0] = *(const v8bf*)(qr + kk * 32 + half * 16);
      qa[qi][kk].h[1] = *(const v8bf*)(qr + kk * 32 + half * 16 + 8);
    }
  }

  v8f o[2][4] = {};
  float mrow[2][8], lrow[2][8];
#pragma unroll
  for (int qi = 0; qi < 2; qi++)
#pragma unroll
    for (int r = 0; r < 8; r++) { mrow[qi][r] = -1e30f; lrow[qi][r] = 0.f; }

  const float scale = 0.125f;                         // 1/sqrt(64)

  for (int j = 0; j < S_LEN; j += 32) {
    v16bf kbm[2][2];
#pragma unroll
    for (int ni = 0; ni < 2; ni++) {
      const bf16_t* kr = Kb + (size_t)(j + ni * 16 + l15) * DHEAD;
#pragma unroll
      for (int kk = 0; kk < 2; kk++)
        kbm[ni][kk] = *(const v16bf*)(kr + kk * 32 + half * 16);
    }
    v8f sc[2][2] = {};
#pragma unroll
    for (int qi = 0; qi < 2; qi++)
#pragma unroll
      for (int ni = 0; ni < 2; ni++) {
        sc[qi][ni] = wmma_bf16(qa[qi][0].v, kbm[ni][0], sc[qi][ni]);
        sc[qi][ni] = wmma_bf16(qa[qi][1].v, kbm[ni][1], sc[qi][ni]);
      }

    // online softmax (reductions stay within 16-lane halves == matrix rows)
#pragma unroll
    for (int qi = 0; qi < 2; qi++)
#pragma unroll
      for (int r = 0; r < 8; r++) {
        float s0 = sc[qi][0][r] * scale;
        float s1 = sc[qi][1][r] * scale;
        float mx = fmaxf(s0, s1);
#pragma unroll
        for (int mk = 8; mk >= 1; mk >>= 1) mx = fmaxf(mx, __shfl_xor(mx, mk, 32));
        const float mnew = fmaxf(mrow[qi][r], mx);
        const float alpha = __expf(mrow[qi][r] - mnew);
        const float p0 = __expf(s0 - mnew);
        const float p1 = __expf(s1 - mnew);
        float rs = p0 + p1;
#pragma unroll
        for (int mk = 8; mk >= 1; mk >>= 1) rs += __shfl_xor(rs, mk, 32);
        lrow[qi][r] = lrow[qi][r] * alpha + rs;
        mrow[qi][r] = mnew;
        sc[qi][0][r] = p0;
        sc[qi][1][r] = p1;
#pragma unroll
        for (int di = 0; di < 4; di++) o[qi][di][r] *= alpha;
      }

    // stage P (C layout) into LDS as bf16
#pragma unroll
    for (int qi = 0; qi < 2; qi++)
#pragma unroll
      for (int ni = 0; ni < 2; ni++)
#pragma unroll
        for (int r = 0; r < 8; r++)
          pst[wave][qi * 16 + half * 8 + r][ni * 16 + l15] = (bf16_t)sc[qi][ni][r];
    asm volatile("s_wait_dscnt 0x0" ::: "memory");

    // reload P as A-operand layout
    V16 pa[2];
#pragma unroll
    for (int qi = 0; qi < 2; qi++) {
      const bf16_t* pr = &pst[wave][qi * 16 + l15][0];
      pa[qi].h[0] = *(const v8bf*)(pr + half * 16);
      pa[qi].h[1] = *(const v8bf*)(pr + half * 16 + 8);
    }
    // V operand (B role): contiguous over key positions in Vt
    v16bf vbt[4];
#pragma unroll
    for (int di = 0; di < 4; di++) {
      const bf16_t* vr = Vb + (size_t)(di * 16 + l15) * S_LEN + j;
      vbt[di] = *(const v16bf*)(vr + half * 16);
    }
#pragma unroll
    for (int qi = 0; qi < 2; qi++)
#pragma unroll
      for (int di = 0; di < 4; di++)
        o[qi][di] = wmma_bf16(pa[qi].v, vbt[di], o[qi][di]);
  }

  // epilogue: divide by l, write bf16 to [S][D] for the output projection
#pragma unroll
  for (int qi = 0; qi < 2; qi++)
#pragma unroll
    for (int r = 0; r < 8; r++) {
      const float inv = 1.0f / lrow[qi][r];
      const int row = q0 + qi * 16 + half * 8 + r;
#pragma unroll
      for (int di = 0; di < 4; di++) {
        const int col = h * DHEAD + di * 16 + l15;
        Out[(size_t)row * D_MODEL + col] = (bf16_t)(o[qi][di][r] * inv);
      }
    }
}

// ---------------------------------------------------------------------------
// Host orchestration
// ---------------------------------------------------------------------------
extern "C" void kernel_launch(void* const* d_in, const int* in_sizes, int n_in,
                              void* d_out, int out_size, void* d_ws, size_t ws_size,
                              hipStream_t stream) {
  (void)in_sizes; (void)n_in; (void)out_size; (void)ws_size;
  const float* X    = (const float*)d_in[0];
  const float* cosT = (const float*)d_in[1];
  const float* sinT = (const float*)d_in[2];
  const float* Wq   = (const float*)d_in[3];
  const float* bq   = (const float*)d_in[4];
  const float* Wk   = (const float*)d_in[5];
  const float* bk   = (const float*)d_in[6];
  const float* Wv   = (const float*)d_in[7];
  const float* bv   = (const float*)d_in[8];
  const float* qw   = (const float*)d_in[9];
  const float* kw   = (const float*)d_in[10];
  const float* Wo   = (const float*)d_in[11];
  const float* bo   = (const float*)d_in[12];
  float* out = (float*)d_out;

  const size_t SZ = (size_t)S_LEN * D_MODEL;   // 4M elems
  char* p = (char*)d_ws;
  auto take = [&](size_t bytes) -> void* {
    void* r = (void*)p; p += (bytes + 255) & ~(size_t)255; return r;
  };
  bf16_t* Xb  = (bf16_t*)take(SZ * 2);
  bf16_t* WqT = (bf16_t*)take(SZ * 2);
  bf16_t* WkT = (bf16_t*)take(SZ * 2);
  bf16_t* WvT = (bf16_t*)take(SZ * 2);
  bf16_t* WoT = (bf16_t*)take(SZ * 2);
  float*  Qf  = (float*) take(SZ * 4);
  float*  Kf  = (float*) take(SZ * 4);
  float*  Vf  = (float*) take(SZ * 4);
  bf16_t* QhB = (bf16_t*)take(SZ * 2);
  bf16_t* KhB = (bf16_t*)take(SZ * 2);
  bf16_t* VtB = (bf16_t*)take(SZ * 2);
  bf16_t* AoB = (bf16_t*)take(SZ * 2);

  const dim3 b256(256);
  k_cvt_bf16<<<dim3((unsigned)(SZ / 1024)), b256, 0, stream>>>(X, Xb, (int)(SZ / 4));

  const dim3 tb(32, 8), tg(64, 64);
  k_transpose_cvt<<<tg, tb, 0, stream>>>(Wq, WqT);
  k_transpose_cvt<<<tg, tb, 0, stream>>>(Wk, WkT);
  k_transpose_cvt<<<tg, tb, 0, stream>>>(Wv, WvT);
  k_transpose_cvt<<<tg, tb, 0, stream>>>(Wo, WoT);

  const dim3 gg(D_MODEL / 128, S_LEN / 128);   // (16,16)
  k_gemm_bf16<<<gg, b256, 0, stream>>>(Xb, WqT, bq, Qf, S_LEN, D_MODEL, D_MODEL);
  k_gemm_bf16<<<gg, b256, 0, stream>>>(Xb, WkT, bk, Kf, S_LEN, D_MODEL, D_MODEL);
  k_gemm_bf16<<<gg, b256, 0, stream>>>(Xb, WvT, bv, Vf, S_LEN, D_MODEL, D_MODEL);

  k_rmsnorm_rope<<<dim3(S_LEN), b256, 0, stream>>>(Qf, qw, cosT, sinT, QhB);
  k_rmsnorm_rope<<<dim3(S_LEN), b256, 0, stream>>>(Kf, kw, cosT, sinT, KhB);
  k_reorder_v<<<dim3((unsigned)(SZ / 256)), b256, 0, stream>>>(Vf, VtB);

  k_attention<<<dim3(NHEAD, S_LEN / 256), b256, 0, stream>>>(QhB, KhB, VtB, AoB);

  k_gemm_bf16<<<gg, b256, 0, stream>>>(AoB, WoT, bo, out, S_LEN, D_MODEL, D_MODEL);
}
